// SelfAttention_32744830665567
// MI455X (gfx1250) — matrix operand order repaired and checked by reading, not compiled, not run on hardware
//
#include <hip/hip_runtime.h>

// ---------------------------------------------------------------------------
// Self-attention (faithful to the reference's head-mixing quirks) for
// MI455X / gfx1250.  Heavy math in v_wmma_f32_16x16x32_bf16 (fp32 accum),
// GEMM tile staging via the Tensor Data Mover (tensor_load_to_lds) when the
// toolchain exposes it, with a transpose-free manual fallback.
// ---------------------------------------------------------------------------

typedef __attribute__((ext_vector_type(16))) __bf16        v16bf;
typedef __attribute__((ext_vector_type(8)))  float         v8f;
typedef __attribute__((ext_vector_type(4)))  unsigned int  u32x4;
typedef __attribute__((ext_vector_type(8)))  int           i32x8;
typedef __attribute__((ext_vector_type(4)))  int           i32x4;

#if __has_builtin(__builtin_amdgcn_tensor_load_to_lds) && \
    __has_builtin(__builtin_amdgcn_s_wait_tensorcnt)
#define USE_TDM 1
#else
#define USE_TDM 0
#endif

union Frag16 {
    uint4          q[2];
    unsigned short s[16];
    v16bf          v;
};

__device__ __forceinline__ unsigned short f2bf(float f) {
    union { float f; unsigned u; } v; v.f = f;
    unsigned r = v.u + 0x7FFFu + ((v.u >> 16) & 1u);   // round-to-nearest-even
    return (unsigned short)(r >> 16);
}

__device__ __forceinline__ unsigned lds_off32(const void* p) {
    // flat LDS address = {aperture, offset[31:0]} -> low 32 bits are the offset
    return (unsigned)(uintptr_t)p;
}

#if USE_TDM
// One TDM descriptor issue: copy a 128-row x 64-byte tile (row stride 2048 B)
// from global memory into LDS, contiguous.  data_size = 8 B units.
// This toolchain exposes the 6-arg builtin:
//   (u32x4 g0, i32x8 g1, i32x4 g2, i32x4 g3, i32x8 extra, i32 cpol)
__device__ __forceinline__ void tdm_tile_128x64B(const void* gsrc, unsigned ldsoff) {
    union { u32x4 v; unsigned w[4]; } g0;
    unsigned long long ga = (unsigned long long)(uintptr_t)gsrc;
    g0.w[0] = 1u;                                   // count=1 (valid), no gather
    g0.w[1] = ldsoff;                               // lds_addr
    g0.w[2] = (unsigned)ga;                         // global_addr[31:0]
    g0.w[3] = (unsigned)((ga >> 32) & 0x1FFFFFFu)   // global_addr[56:32]
            | (2u << 30);                           // type = 2 ("image")
    union { i32x8 v; int w[8]; } g1;
    g1.w[0] = 3 << 16;        // wg_mask=0 | data_size=3 (8B) | no flags
    g1.w[1] = 8 << 16;        // tensor_dim0 = 8 units   (bits 79:48, low half)
    g1.w[2] = 128 << 16;      // tensor_dim0 hi=0 | tensor_dim1 = 128 (low half)
    g1.w[3] = 8 << 16;        // tensor_dim1 hi=0 | tile_dim0 = 8 units
    g1.w[4] = 128;            // tile_dim1 = 128 | tile_dim2 = 0
    g1.w[5] = 256;            // tensor_dim0_stride = 256 units (2048 B)
    g1.w[6] = 0;              // stride hi | tensor_dim1_stride lo (unused, 2D)
    g1.w[7] = 0;
    i32x4 z4 = {0, 0, 0, 0};                        // groups 2/3 unused (<=2D)
    i32x8 z8 = {0, 0, 0, 0, 0, 0, 0, 0};            // unused trailing group
    __builtin_amdgcn_tensor_load_to_lds(g0.v, g1.v, z4, z4, z8, 0);
}
#endif

// ---------------------------------------------------------------------------
// fp32 -> bf16 conversion, 4 elements / thread (pure bandwidth)
// ---------------------------------------------------------------------------
__global__ __launch_bounds__(256) void cvt_f32_to_bf16(
    const float* __restrict__ in, unsigned short* __restrict__ out, int n4)
{
    int i = blockIdx.x * blockDim.x + threadIdx.x;
    if (i >= n4) return;
    float4 f = ((const float4*)in)[i];
    union { unsigned short s[4]; uint2 u; } r;
    r.s[0] = f2bf(f.x); r.s[1] = f2bf(f.y);
    r.s[2] = f2bf(f.z); r.s[3] = f2bf(f.w);
    ((uint2*)out)[i] = r.u;
}

// ---------------------------------------------------------------------------
// fp32 [K,N] -> bf16 transposed [N,K], 64x64 tiles through LDS.
// One-time cost per weight matrix; makes every GEMM tile K-contiguous so the
// hot loop never transposes anything.
// ---------------------------------------------------------------------------
__global__ __launch_bounds__(256) void cvt_transpose_bf16(
    const float* __restrict__ W,        // [1024,1024] fp32
    unsigned short* __restrict__ Wt)    // [1024,1024] bf16, transposed
{
    __shared__ unsigned short tile[64][68];          // [n_local][k_local], padded
    const int k0 = blockIdx.y * 64, n0 = blockIdx.x * 64;
    const int tx = threadIdx.x & 15, ty = threadIdx.x >> 4;
#pragma unroll
    for (int i = 0; i < 4; ++i) {
        const int kk = ty + i * 16;
        float4 f = *(const float4*)&W[(size_t)(k0 + kk) * 1024 + n0 + tx * 4];
        tile[tx * 4 + 0][kk] = f2bf(f.x);
        tile[tx * 4 + 1][kk] = f2bf(f.y);
        tile[tx * 4 + 2][kk] = f2bf(f.z);
        tile[tx * 4 + 3][kk] = f2bf(f.w);
    }
    __syncthreads();
#pragma unroll
    for (int i = 0; i < 4; ++i) {
        const int nl = ty + i * 16;
        union { unsigned short s[4]; uint2 u; } r;
        r.s[0] = tile[nl][tx * 4 + 0];
        r.s[1] = tile[nl][tx * 4 + 1];
        r.s[2] = tile[nl][tx * 4 + 2];
        r.s[3] = tile[nl][tx * 4 + 3];
        *(uint2*)&Wt[(size_t)(n0 + nl) * 1024 + k0 + tx * 4] = r.u;
    }
}

// ---------------------------------------------------------------------------
// bf16 GEMM: C[M,N] = A[M,K] @ Wt[N,K]^T + bias[N]
//   block tile 128x128, BK=32, 8 waves, wave tile 32x64 (2x4 16x16 WMMAs)
//   double-buffered LDS; both tiles staged by TDM (one descriptor each).
// ---------------------------------------------------------------------------
template <bool OUT_BF16>
__global__ __launch_bounds__(256) void gemm_bf16_wmma(
    const unsigned short* __restrict__ A,    // [M,K] bf16
    const unsigned short* __restrict__ Wt,   // [N,K] bf16 (pre-transposed)
    const float* __restrict__ bias,          // [N]   fp32
    void* __restrict__ C,                    // [M,N] bf16 or fp32
    int M, int N, int K)
{
    constexpr int BM = 128, BN = 128, BK = 32;
    __shared__ unsigned short As[2][BM][BK];   // 16 KB
    __shared__ unsigned short Bs[2][BN][BK];   // 16 KB ([n][k])

    const int tid  = threadIdx.x;
    const int wave = tid >> 5, lane = tid & 31;
    const int kh   = lane >> 4, lr = lane & 15;     // lane-half, lane-row
    const int wm   = wave & 3,  wn = wave >> 2;     // 4x2 wave grid
    const int m0   = blockIdx.y * BM, n0 = blockIdx.x * BN;

    auto stage = [&](int k0, int buf) {
#if USE_TDM
        if (tid == 0) {
            tdm_tile_128x64B(A  + (size_t)m0 * K + k0, lds_off32(&As[buf][0][0]));
            tdm_tile_128x64B(Wt + (size_t)n0 * K + k0, lds_off32(&Bs[buf][0][0]));
        }
#else
        const int r = tid >> 1, seg = (tid & 1) * 16;      // 2 threads / row
        const unsigned short* pa = A + (size_t)(m0 + r) * K + k0 + seg;
        __builtin_prefetch((const void*)(pa + 64), 0, 0);  // global_prefetch_b8
        uint4 a0 = ((const uint4*)pa)[0];
        uint4 a1 = ((const uint4*)pa)[1];
        *((uint4*)&As[buf][r][seg])     = a0;
        *((uint4*)&As[buf][r][seg + 8]) = a1;
        const unsigned short* pb = Wt + (size_t)(n0 + r) * K + k0 + seg;
        __builtin_prefetch((const void*)(pb + 64), 0, 0);
        uint4 b0 = ((const uint4*)pb)[0];
        uint4 b1 = ((const uint4*)pb)[1];
        *((uint4*)&Bs[buf][r][seg])     = b0;
        *((uint4*)&Bs[buf][r][seg + 8]) = b1;
#endif
    };
    auto publish = [&]() {            // make staged buffer visible to block
#if USE_TDM
        if (wave == 0) __builtin_amdgcn_s_wait_tensorcnt(0);
#endif
        __syncthreads();
    };

    v8f acc[2][4];
#pragma unroll
    for (int i = 0; i < 2; ++i)
#pragma unroll
        for (int j = 0; j < 4; ++j) acc[i][j] = (v8f){0,0,0,0,0,0,0,0};

    stage(0, 0);
    publish();

    const int KSTEPS = K / BK;
    for (int ks = 0; ks < KSTEPS; ++ks) {
        const int cur = ks & 1;
        if (ks + 1 < KSTEPS) stage((ks + 1) * BK, cur ^ 1);

        // A-frag (ISA 16-bit A layout: halves = kh*8+0..7, 16+kh*8+0..7)
        Frag16 af[2];
#pragma unroll
        for (int mt = 0; mt < 2; ++mt) {
            const int m = wm * 32 + mt * 16 + lr;
            af[mt].q[0] = *((const uint4*)&As[cur][m][kh * 8]);
            af[mt].q[1] = *((const uint4*)&As[cur][m][16 + kh * 8]);
        }
        // B-frag (lane=N, halves = K = kh*8+0..7, 16+kh*8+0..7, matching A)
        Frag16 bfr[4];
#pragma unroll
        for (int nt = 0; nt < 4; ++nt) {
            const int n = wn * 64 + nt * 16 + lr;
            bfr[nt].q[0] = *((const uint4*)&Bs[cur][n][kh * 8]);
            bfr[nt].q[1] = *((const uint4*)&Bs[cur][n][16 + kh * 8]);
        }
#pragma unroll
        for (int mt = 0; mt < 2; ++mt)
#pragma unroll
            for (int nt = 0; nt < 4; ++nt)
                acc[mt][nt] = __builtin_amdgcn_wmma_f32_16x16x32_bf16(
                    false, af[mt].v, false, bfr[nt].v,
                    (short)0, acc[mt][nt], false, false);
        publish();
    }

    // epilogue: C layout -> VGPR r holds row (r + kh*8), col = lr
#pragma unroll
    for (int mt = 0; mt < 2; ++mt) {
#pragma unroll
        for (int nt = 0; nt < 4; ++nt) {
            const int col = n0 + wn * 64 + nt * 16 + lr;
            const float bia = bias[col];
#pragma unroll
            for (int r = 0; r < 8; ++r) {
                const int row = m0 + wm * 32 + mt * 16 + kh * 8 + r;
                const float val = acc[mt][nt][r] + bia;
                if (OUT_BF16)
                    ((unsigned short*)C)[(size_t)row * N + col] = f2bf(val);
                else
                    ((float*)C)[(size_t)row * N + col] = val;
            }
        }
    }
}

// ---------------------------------------------------------------------------
// Per-token attention: scores = (q @ k^T)/8 over heads (16x16, K=64),
// softmax over the head axis, out = attn @ v.  One wave per token.
// Output written directly in the reference's scrambled [B][H][S][DK] layout.
// ---------------------------------------------------------------------------
__global__ __launch_bounds__(256) void attn_softmax_wmma(
    const unsigned short* __restrict__ Q,
    const unsigned short* __restrict__ Kb,
    const unsigned short* __restrict__ V,
    unsigned short* __restrict__ O)     // [B][H][S][DK] bf16
{
    __shared__ unsigned short At[8][16][16];   // attn matrix per wave, 4 KB

    const int tid  = threadIdx.x;
    const int wave = tid >> 5, lane = tid & 31;
    const int kh   = lane >> 4, lr = lane & 15;
    const int tok  = blockIdx.x * 8 + wave;    // 0..16383
    const int b    = tok >> 12, s = tok & 4095;

    const unsigned short* q = Q  + (size_t)tok * 1024;
    const unsigned short* k = Kb + (size_t)tok * 1024;
    const unsigned short* v = V  + (size_t)tok * 1024;

    // ---- scores: two chained 16x16x32 bf16 WMMAs over DK=64 ----
    v8f sc = (v8f){0,0,0,0,0,0,0,0};
#pragma unroll
    for (int c = 0; c < 2; ++c) {
        Frag16 aq, bk;
        const unsigned short* qr = q + lr * 64 + c * 32;
        aq.q[0] = *((const uint4*)(qr + kh * 8));
        aq.q[1] = *((const uint4*)(qr + 16 + kh * 8));
        const unsigned short* kr = k + lr * 64 + c * 32;   // B = k^T
        bk.q[0] = *((const uint4*)(kr + kh * 8));
        bk.q[1] = *((const uint4*)(kr + 16 + kh * 8));
        sc = __builtin_amdgcn_wmma_f32_16x16x32_bf16(
            false, aq.v, false, bk.v, (short)0, sc, false, false);
    }

    // ---- softmax over N (16 lanes of this half-wave), row = kh*8 + r ----
#pragma unroll
    for (int r = 0; r < 8; ++r) {
        const float xv = sc[r] * 0.125f;               // 1/sqrt(64)
        float mx = xv;
        mx = fmaxf(mx, __shfl_xor(mx, 1, 32));
        mx = fmaxf(mx, __shfl_xor(mx, 2, 32));
        mx = fmaxf(mx, __shfl_xor(mx, 4, 32));
        mx = fmaxf(mx, __shfl_xor(mx, 8, 32));
        const float e = __expf(xv - mx);
        float sm = e;
        sm += __shfl_xor(sm, 1, 32);
        sm += __shfl_xor(sm, 2, 32);
        sm += __shfl_xor(sm, 4, 32);
        sm += __shfl_xor(sm, 8, 32);
        At[wave][kh * 8 + r][lr] = f2bf(e / sm);       // re-layout via LDS
    }
    __syncthreads();

    // ---- out = attn(16x16) @ v(16x64): K padded 16->32 with zeros ----
    Frag16 ap;
    ap.q[0] = *((const uint4*)&At[wave][lr][kh * 8]);
    uint4 z4; z4.x = z4.y = z4.z = z4.w = 0u;
    ap.q[1] = z4;                                      // K = 16..31 are zero

#pragma unroll
    for (int nt = 0; nt < 4; ++nt) {
        Frag16 fb;                                     // B[j][d], j = kh*8+i (i<8)
#pragma unroll
        for (int i = 0; i < 16; ++i) {
            unsigned short t = v[(size_t)(kh * 8 + (i & 7)) * 64 + nt * 16 + lr];
            fb.s[i] = (i >= 8) ? (unsigned short)0 : t; // j >= 16 -> zero pad
        }
        v8f zero = (v8f){0,0,0,0,0,0,0,0};
        v8f o = __builtin_amdgcn_wmma_f32_16x16x32_bf16(
            false, ap.v, false, fb.v, (short)0, zero, false, false);
#pragma unroll
        for (int r = 0; r < 8; ++r) {
            const int head = kh * 8 + r;
            O[(((size_t)b * 16 + head) * 4096 + s) * 64 + nt * 16 + lr] =
                f2bf(o[r]);
        }
    }
}

// ---------------------------------------------------------------------------
extern "C" void kernel_launch(void* const* d_in, const int* in_sizes, int n_in,
                              void* d_out, int out_size, void* d_ws, size_t ws_size,
                              hipStream_t stream)
{
    const float* x  = (const float*)d_in[0];
    const float* Wq = (const float*)d_in[1];
    const float* bq = (const float*)d_in[2];
    const float* Wk = (const float*)d_in[3];
    const float* bk = (const float*)d_in[4];
    const float* Wv = (const float*)d_in[5];
    const float* bv = (const float*)d_in[6];
    const float* Wo = (const float*)d_in[7];
    const float* bo = (const float*)d_in[8];

    const size_t MTOK = 16384, DM = 1024;      // B*S rows, model dim

    char* p = (char*)d_ws;
    unsigned short* xb  = (unsigned short*)p; p += MTOK * DM * 2;   // 32 MB
    unsigned short* wqb = (unsigned short*)p; p += DM * DM * 2;     //  2 MB (transposed)
    unsigned short* wkb = (unsigned short*)p; p += DM * DM * 2;
    unsigned short* wvb = (unsigned short*)p; p += DM * DM * 2;
    unsigned short* wob = (unsigned short*)p; p += DM * DM * 2;
    unsigned short* qb  = (unsigned short*)p; p += MTOK * DM * 2;   // 32 MB
    unsigned short* kb2 = (unsigned short*)p; p += MTOK * DM * 2;
    unsigned short* vb2 = (unsigned short*)p; p += MTOK * DM * 2;
    unsigned short* ab  = (unsigned short*)p; p += MTOK * DM * 2;   // scrambled

    // activations: plain fp32 -> bf16
    cvt_f32_to_bf16<<<(int)(MTOK * DM / 4 / 256), 256, 0, stream>>>(
        x, xb, (int)(MTOK * DM / 4));
    // weights: fp32 [K,N] -> bf16 [N,K] so GEMM tiles are K-contiguous
    dim3 tg(16, 16);
    cvt_transpose_bf16<<<tg, 256, 0, stream>>>(Wq, wqb);
    cvt_transpose_bf16<<<tg, 256, 0, stream>>>(Wk, wkb);
    cvt_transpose_bf16<<<tg, 256, 0, stream>>>(Wv, wvb);
    cvt_transpose_bf16<<<tg, 256, 0, stream>>>(Wo, wob);

    // Q/K/V projections (bf16 out for the attention stage)
    dim3 gg(1024 / 128, 16384 / 128);          // (8, 128)
    gemm_bf16_wmma<true><<<gg, 256, 0, stream>>>(xb, wqb, bq, qb,  16384, 1024, 1024);
    gemm_bf16_wmma<true><<<gg, 256, 0, stream>>>(xb, wkb, bk, kb2, 16384, 1024, 1024);
    gemm_bf16_wmma<true><<<gg, 256, 0, stream>>>(xb, wvb, bv, vb2, 16384, 1024, 1024);

    // per-token head-vs-head attention, writes scrambled [B][H][S][DK]
    attn_softmax_wmma<<<16384 / 8, 256, 0, stream>>>(qb, kb2, vb2, ab);

    // output projection: reads scrambled buffer flat as [16384, 1024]
    gemm_bf16_wmma<false><<<gg, 256, 0, stream>>>(ab, wob, bo, d_out, 16384, 1024, 1024);
}
